// GAT_79980880986112
// MI455X (gfx1250) — matrix operand order repaired and checked by reading, not compiled, not run on hardware
//
#include <hip/hip_runtime.h>
#include <math.h>

// ---------- types ----------
typedef __attribute__((ext_vector_type(16))) __bf16       v16bf;
typedef __attribute__((ext_vector_type(8)))  float        v8f;
typedef __attribute__((ext_vector_type(8)))  unsigned int v8u;

__device__ __forceinline__ unsigned short f2bf(float f) {
    unsigned int u = __float_as_uint(f);
    u += 0x7FFFu + ((u >> 16) & 1u);      // round-to-nearest-even
    return (unsigned short)(u >> 16);
}

__device__ __forceinline__ void atomicMaxF(float* addr, float val) {
    // IEEE-754 bit-pattern ordering trick: works for mixed signs
    if (val >= 0.0f) atomicMax((int*)addr, __float_as_int(val));
    else             atomicMin((unsigned int*)addr, __float_as_uint(val));
}

// =====================================================================
// bf16 WMMA GEMM:  C[M,Nout] = A[M,K] * B[K,Nout]   (A,B,C row-major f32)
// Block = 256 threads (8 waves). Block tile = 128 rows x 64 cols.
// Wave w owns rows [w*16, w*16+16) and 4 column tiles of 16.
// A fragment: built per-lane straight from global (lane = M%16 + 16*Khalf).
// B fragment: staged once per K-step into LDS in WMMA fragment order.
// K must be a multiple of 32 (true here: 256/128).
// =====================================================================
__global__ __launch_bounds__(256) void gemm_bf16_wmma(
        const float* __restrict__ A, const float* __restrict__ B,
        float* __restrict__ C, int M, int K, int Nout) {
    __shared__ __attribute__((aligned(32))) unsigned short Bfrag[4 * 32 * 16];

    const int t  = threadIdx.x;
    const int w  = t >> 5;       // wave id 0..7
    const int l  = t & 31;       // lane
    const int li = l & 15;       // M (or N) within tile
    const int hi = l >> 4;       // K-half selector
    const int rowBase = blockIdx.x * 128;
    const int col0    = blockIdx.y * 64;

    int mA = rowBase + w * 16 + li;
    if (mA > M - 1) mA = M - 1;                 // clamp loads; stores are guarded
    const float* arow = A + (size_t)mA * K;

    // B-loader assignment: 128 fragment-lanes (4 tiles x 32 lanes), 2 threads each
    const int f   = t >> 1;                     // fragment-lane 0..127
    const int hh  = t & 1;                      // element half (8 elems each)
    const int bl  = f & 31;
    const int bn  = col0 + (f >> 5) * 16 + (bl & 15);
    const int bhi = bl >> 4;

    v8f acc0 = {}, acc1 = {}, acc2 = {}, acc3 = {};

    for (int kb = 0; kb < K; kb += 32) {
        // ---- stage B tile into LDS in fragment order ----
        {
            const int eb = hh * 8;
            #pragma unroll
            for (int j = 0; j < 8; ++j) {
                const int e = eb + j;
                const int k = kb + 8 * bhi + e + eb;      // B elem e -> K = 8*hi + e (e<8), 16 + 8*hi + (e-8) (e>=8)
                Bfrag[f * 16 + e] = f2bf(B[(size_t)k * Nout + bn]);
            }
        }
        // ---- A fragment straight to registers (2x contiguous float8) ----
        union { v16bf v; unsigned short s[16]; } af;
        {
            const int k0 = kb + 8 * hi;                   // e<8  -> K = 8*hi + e
            const int k1 = k0 + 16;                       // e>=8 -> K = 16 + 8*hi + (e-8)
            float4 a0 = *(const float4*)(arow + k0);
            float4 a1 = *(const float4*)(arow + k0 + 4);
            float4 a2 = *(const float4*)(arow + k1);
            float4 a3 = *(const float4*)(arow + k1 + 4);
            af.s[0]  = f2bf(a0.x); af.s[1]  = f2bf(a0.y); af.s[2]  = f2bf(a0.z); af.s[3]  = f2bf(a0.w);
            af.s[4]  = f2bf(a1.x); af.s[5]  = f2bf(a1.y); af.s[6]  = f2bf(a1.z); af.s[7]  = f2bf(a1.w);
            af.s[8]  = f2bf(a2.x); af.s[9]  = f2bf(a2.y); af.s[10] = f2bf(a2.z); af.s[11] = f2bf(a2.w);
            af.s[12] = f2bf(a3.x); af.s[13] = f2bf(a3.y); af.s[14] = f2bf(a3.z); af.s[15] = f2bf(a3.w);
        }
        __builtin_prefetch(arow + kb + 32, 0, 0);         // global_prefetch_b8
        __syncthreads();

        union { v8u u; v16bf v; } b0, b1, b2, b3;
        b0.u = *(const v8u*)&Bfrag[(0 * 32 + l) * 16];
        b1.u = *(const v8u*)&Bfrag[(1 * 32 + l) * 16];
        b2.u = *(const v8u*)&Bfrag[(2 * 32 + l) * 16];
        b3.u = *(const v8u*)&Bfrag[(3 * 32 + l) * 16];

        acc0 = __builtin_amdgcn_wmma_f32_16x16x32_bf16(false, af.v, false, b0.v, (short)0, acc0, false, false);
        acc1 = __builtin_amdgcn_wmma_f32_16x16x32_bf16(false, af.v, false, b1.v, (short)0, acc1, false, false);
        acc2 = __builtin_amdgcn_wmma_f32_16x16x32_bf16(false, af.v, false, b2.v, (short)0, acc2, false, false);
        acc3 = __builtin_amdgcn_wmma_f32_16x16x32_bf16(false, af.v, false, b3.v, (short)0, acc3, false, false);
        __syncthreads();
    }

    // C/D layout: lane l -> column li ; VGPR r -> row r + 8*hi
    #pragma unroll
    for (int r = 0; r < 8; ++r) {
        const int mo = rowBase + w * 16 + r + 8 * hi;
        if (mo < M) {
            float* crow = C + (size_t)mo * Nout + col0 + li;
            crow[0]  = acc0[r];
            crow[16] = acc1[r];
            crow[32] = acc2[r];
            crow[48] = acc3[r];
        }
    }
}

// ---------------------------------------------------------------------
__global__ void zero_f32(float* __restrict__ p, long n) {
    long g = (long)blockIdx.x * blockDim.x + threadIdx.x;
    if (g < n) p[g] = 0.0f;
}

// el/er per (node, head); init softmax running max/sum
__global__ void node_prep(const float* __restrict__ hfeat,
                          const float* __restrict__ al, const float* __restrict__ ar,
                          float* __restrict__ el, float* __restrict__ er,
                          float* __restrict__ mx, float* __restrict__ sm,
                          int N, int H, int D) {
    long g = (long)blockIdx.x * blockDim.x + threadIdx.x;
    if (g >= (long)N * H) return;
    const int n = (int)(g / H), h = (int)(g % H);
    const float* row = hfeat + (long)n * H * D + h * D;
    const float* alr = al + h * D;
    const float* arr = ar + h * D;
    float accl = 0.f, accr = 0.f;
    for (int d = 0; d < D; d += 4) {
        float4 v = *(const float4*)(row + d);
        float4 a = *(const float4*)(alr + d);
        float4 r = *(const float4*)(arr + d);
        accl += v.x * a.x + v.y * a.y + v.z * a.z + v.w * a.w;
        accr += v.x * r.x + v.y * r.y + v.z * r.z + v.w * r.w;
    }
    el[g] = accl; er[g] = accr; mx[g] = -1e30f; sm[g] = 0.0f;
}

__global__ void edge_max(const int* __restrict__ src, const int* __restrict__ dst,
                         const float* __restrict__ el, const float* __restrict__ er,
                         float* __restrict__ mx, int E, int H) {
    int e = blockIdx.x * blockDim.x + threadIdx.x;
    if (e >= E) return;
    const int s = src[e], d = dst[e];
    for (int h = 0; h < H; ++h) {
        float v = el[s * H + h] + er[d * H + h];
        v = (v > 0.f) ? v : 0.2f * v;                 // leaky_relu
        atomicMaxF(&mx[d * H + h], v);
    }
}

__global__ void edge_expsum(const int* __restrict__ src, const int* __restrict__ dst,
                            const float* __restrict__ el, const float* __restrict__ er,
                            const float* __restrict__ mx,
                            float* __restrict__ exb, float* __restrict__ sm, int E, int H) {
    int e = blockIdx.x * blockDim.x + threadIdx.x;
    if (e >= E) return;
    const int s = src[e], d = dst[e];
    for (int h = 0; h < H; ++h) {
        float v = el[s * H + h] + er[d * H + h];
        v = (v > 0.f) ? v : 0.2f * v;
        float ex = __expf(v - mx[d * H + h]);
        exb[(long)e * H + h] = ex;
        atomicAdd(&sm[d * H + h], ex);
    }
}

// one float4 chunk per thread; HD/4 threads per edge
__global__ void edge_scatter(const int* __restrict__ src, const int* __restrict__ dst,
                             const float* __restrict__ hfeat,
                             const float* __restrict__ exb, const float* __restrict__ sm,
                             float* __restrict__ agg, int E, int H, int D) {
    const int HD = H * D, cpe = HD / 4;
    long g = (long)blockIdx.x * blockDim.x + threadIdx.x;
    if (g >= (long)E * cpe) return;
    const int e = (int)(g / cpe), c = (int)(g % cpe);
    const int s = src[e], d = dst[e];
    const int h = (c * 4) / D;
    const float alpha = exb[(long)e * H + h] / sm[d * H + h];
    float4 hv = *(const float4*)(hfeat + (long)s * HD + c * 4);
    float* o = agg + (long)d * HD + c * 4;
    atomicAdd(o + 0, hv.x * alpha);
    atomicAdd(o + 1, hv.y * alpha);
    atomicAdd(o + 2, hv.z * alpha);
    atomicAdd(o + 3, hv.w * alpha);
}

// out = elu(agg + b [+ resid]); resid may be nullptr; out may alias resid (elementwise)
__global__ void finalize_bias_elu(const float* __restrict__ agg, const float* __restrict__ b,
                                  const float* resid, float* __restrict__ out, int N, int HD) {
    long g = (long)blockIdx.x * blockDim.x + threadIdx.x;
    if (g >= (long)N * HD) return;
    const int j = (int)(g % HD);
    float v = agg[g] + b[j];
    if (resid) v += resid[g];
    out[g] = (v > 0.f) ? v : (__expf(v) - 1.0f);
}

// layer-2 epilogue: projected residual + bias, then mean over H=2 heads
__global__ void head_mean(const float* __restrict__ agg, const float* __restrict__ res,
                          const float* __restrict__ b, float* __restrict__ out,
                          int N, int CLS) {
    long g = (long)blockIdx.x * blockDim.x + threadIdx.x;
    if (g >= (long)N * CLS) return;
    const int n = (int)(g / CLS), c = (int)(g % CLS);
    const int HD = 2 * CLS;
    float v0 = agg[(long)n * HD + c]       + res[(long)n * HD + c]       + b[c];
    float v1 = agg[(long)n * HD + CLS + c] + res[(long)n * HD + CLS + c] + b[CLS + c];
    out[g] = 0.5f * (v0 + v1);
}

// =====================================================================
extern "C" void kernel_launch(void* const* d_in, const int* in_sizes, int n_in,
                              void* d_out, int out_size, void* d_ws, size_t ws_size,
                              hipStream_t stream) {
    const float* x     = (const float*)d_in[0];
    const int*   esrc  = (const int*)d_in[1];
    const int*   edst  = (const int*)d_in[2];
    const float* W0    = (const float*)d_in[3];
    const float* al0   = (const float*)d_in[4];
    const float* ar0   = (const float*)d_in[5];
    const float* b0    = (const float*)d_in[6];
    const float* W1    = (const float*)d_in[7];
    const float* al1   = (const float*)d_in[8];
    const float* ar1   = (const float*)d_in[9];
    const float* b1    = (const float*)d_in[10];
    const float* W2    = (const float*)d_in[11];
    const float* al2   = (const float*)d_in[12];
    const float* ar2   = (const float*)d_in[13];
    const float* b2    = (const float*)d_in[14];
    const float* Wres2 = (const float*)d_in[15];

    const int IN = 256, HID = 64, CLS = 32, H = 2;
    const int N = in_sizes[0] / IN;
    const int E = in_sizes[1];

    float* ws   = (float*)d_ws;
    float* hbuf = ws;                              // [N,128]
    float* agg  = hbuf + (size_t)N * 128;          // [N,128]
    float* feat = agg  + (size_t)N * 128;          // [N,128]
    float* res  = feat + (size_t)N * 128;          // [N,64]
    float* el   = res  + (size_t)N * 64;           // [N,H]
    float* er   = el   + (size_t)N * H;
    float* mx   = er   + (size_t)N * H;
    float* sm   = mx   + (size_t)N * H;
    float* exb  = sm   + (size_t)N * H;            // [E,H]

    dim3 blk(256);
    auto cdiv = [](long a, long b) { return (unsigned)((a + b - 1) / b); };

    // -------------------- layer 0 --------------------
    {
        const int K = IN, Nout = H * HID;          // 256 -> 128
        dim3 g(cdiv(N, 128), Nout / 64);
        gemm_bf16_wmma<<<g, blk, 0, stream>>>(x, W0, hbuf, N, K, Nout);
        zero_f32<<<cdiv((long)N * Nout, 256), blk, 0, stream>>>(agg, (long)N * Nout);
        node_prep<<<cdiv((long)N * H, 256), blk, 0, stream>>>(hbuf, al0, ar0, el, er, mx, sm, N, H, HID);
        edge_max<<<cdiv(E, 256), blk, 0, stream>>>(esrc, edst, el, er, mx, E, H);
        edge_expsum<<<cdiv(E, 256), blk, 0, stream>>>(esrc, edst, el, er, mx, exb, sm, E, H);
        edge_scatter<<<cdiv((long)E * (Nout / 4), 256), blk, 0, stream>>>(esrc, edst, hbuf, exb, sm, agg, E, H, HID);
        finalize_bias_elu<<<cdiv((long)N * Nout, 256), blk, 0, stream>>>(agg, b0, nullptr, feat, N, Nout);
    }
    // -------------------- layer 1 (identity residual) --------------------
    {
        const int K = H * HID, Nout = H * HID;     // 128 -> 128
        dim3 g(cdiv(N, 128), Nout / 64);
        gemm_bf16_wmma<<<g, blk, 0, stream>>>(feat, W1, hbuf, N, K, Nout);
        zero_f32<<<cdiv((long)N * Nout, 256), blk, 0, stream>>>(agg, (long)N * Nout);
        node_prep<<<cdiv((long)N * H, 256), blk, 0, stream>>>(hbuf, al1, ar1, el, er, mx, sm, N, H, HID);
        edge_max<<<cdiv(E, 256), blk, 0, stream>>>(esrc, edst, el, er, mx, E, H);
        edge_expsum<<<cdiv(E, 256), blk, 0, stream>>>(esrc, edst, el, er, mx, exb, sm, E, H);
        edge_scatter<<<cdiv((long)E * (Nout / 4), 256), blk, 0, stream>>>(esrc, edst, hbuf, exb, sm, agg, E, H, HID);
        finalize_bias_elu<<<cdiv((long)N * Nout, 256), blk, 0, stream>>>(agg, b1, feat, feat, N, Nout);
    }
    // -------------------- layer 2 (projected residual + head mean) --------------------
    {
        const int K = H * HID, Nout = H * CLS;     // 128 -> 64
        dim3 g(cdiv(N, 128), Nout / 64);
        gemm_bf16_wmma<<<g, blk, 0, stream>>>(feat, W2, hbuf, N, K, Nout);
        gemm_bf16_wmma<<<g, blk, 0, stream>>>(feat, Wres2, res, N, K, Nout);
        zero_f32<<<cdiv((long)N * Nout, 256), blk, 0, stream>>>(agg, (long)N * Nout);
        node_prep<<<cdiv((long)N * H, 256), blk, 0, stream>>>(hbuf, al2, ar2, el, er, mx, sm, N, H, CLS);
        edge_max<<<cdiv(E, 256), blk, 0, stream>>>(esrc, edst, el, er, mx, E, H);
        edge_expsum<<<cdiv(E, 256), blk, 0, stream>>>(esrc, edst, el, er, mx, exb, sm, E, H);
        edge_scatter<<<cdiv((long)E * (Nout / 4), 256), blk, 0, stream>>>(esrc, edst, hbuf, exb, sm, agg, E, H, CLS);
        head_mean<<<cdiv((long)N * CLS, 256), blk, 0, stream>>>(agg, res, b2, (float*)d_out, N, CLS);
    }
}
